// Attention_6476810683049
// MI455X (gfx1250) — matrix operand order repaired and checked by reading, not compiled, not run on hardware
//
#include <hip/hip_runtime.h>
#include <cstdint>
#include <cstddef>

typedef __bf16 bf16;
typedef __attribute__((ext_vector_type(16))) __bf16 bf16x16;
typedef __attribute__((ext_vector_type(8)))  __bf16 bf16x8;
typedef __attribute__((ext_vector_type(8)))  float  f32x8;
typedef __attribute__((ext_vector_type(4)))  unsigned int u32x4;
typedef __attribute__((ext_vector_type(8)))  int i32x8;
typedef __attribute__((ext_vector_type(4)))  int i32x4;

#define DEVINL __device__ __forceinline__

static constexpr int Bz = 2, S = 2048, D = 2048, H = 16, HD = 128;
static constexpr int NQKV = 3 * H * HD; // 6144
static constexpr int BS   = Bz * S;     // 4096

DEVINL bf16 f2bf(float f) {
  uint32_t u = __builtin_bit_cast(uint32_t, f);
  uint32_t r = u + 0x7FFFu + ((u >> 16) & 1u);
  uint16_t h = (uint16_t)(r >> 16);
  return __builtin_bit_cast(bf16, h);
}
DEVINL float bf2f(bf16 b) {
  uint16_t h = __builtin_bit_cast(uint16_t, b);
  uint32_t u = ((uint32_t)h) << 16;
  return __builtin_bit_cast(float, u);
}

// A-operand (16x32 bf16): lane holds row m=lane&15;
// elems 0..15 -> cols 16*half..16*half+15 (matches B operand K order)
DEVINL bf16x16 load_a16x32(const bf16* __restrict__ base, int stride, int lane) {
  const int m = lane & 15, half = lane >> 4;
  const bf16* p = base + m * stride + 16 * half;
  bf16x8 lo = *(const bf16x8*)(p);
  bf16x8 hi = *(const bf16x8*)(p + 8);
  bf16x16 r;
#pragma unroll
  for (int e = 0; e < 8; ++e) { r[e] = lo[e]; r[8 + e] = hi[e]; }
  return r;
}

// B-operand (32x16 bf16): lane holds col n=lane&15, K range 16*(lane>>4)+e.
// Source tile must be laid out so B(k,n) = base[n*stride + k] (k contiguous).
DEVINL bf16x16 load_b32x16(const bf16* __restrict__ base, int stride, int lane) {
  const bf16* p = base + (lane & 15) * stride + (lane >> 4) * 16;
  bf16x8 lo = *(const bf16x8*)(p);
  bf16x8 hi = *(const bf16x8*)(p + 8);
  bf16x16 r;
#pragma unroll
  for (int e = 0; e < 8; ++e) { r[e] = lo[e]; r[8 + e] = hi[e]; }
  return r;
}

DEVINL f32x8 wmma_bf16(bf16x16 a, bf16x16 b, f32x8 c) {
  return __builtin_amdgcn_wmma_f32_16x16x32_bf16(false, a, false, b, (short)0, c,
                                                 false, false);
}

// TDM: DMA a 2D tile of 2-byte elements (global -> LDS).
DEVINL void tdm_load_2d_b16(const void* gptr, uint32_t lds_off,
                            uint32_t dim0, uint32_t dim1,
                            uint32_t tile0, uint32_t tile1, uint64_t stride0) {
  uint64_t ga = (uint64_t)(uintptr_t)gptr;
  u32x4 g0;
  g0[0] = 1u;                                   // count=1 (valid descriptor)
  g0[1] = lds_off;                              // LDS byte address
  g0[2] = (uint32_t)ga;                         // global_addr[31:0]
  g0[3] = (uint32_t)((ga >> 32) & 0x01FFFFFFu)  // global_addr[56:32]
          | 0x80000000u;                        // type=2 ("image")
  i32x8 g1;
  g1[0] = 0x00010000;                           // data_size=1 (2 bytes)
  g1[1] = (int)(dim0 << 16);                    // tensor_dim0[15:0]
  g1[2] = (int)((dim0 >> 16) | (dim1 << 16));   // dim0[31:16] | dim1[15:0]
  g1[3] = (int)((dim1 >> 16) | (tile0 << 16));  // dim1[31:16] | tile_dim0
  g1[4] = (int)tile1;                           // tile_dim1 (tile_dim2=0)
  g1[5] = (int)(uint32_t)stride0;               // tensor_dim0_stride[31:0]
  g1[6] = (int)((stride0 >> 32) & 0xFFFFu);     // stride[47:32] (dim1_stride=0)
  g1[7] = 0;
  i32x4 z4 = {0, 0, 0, 0};
#if defined(__clang_major__) && (__clang_major__ >= 23)
  i32x8 z8 = {0, 0, 0, 0, 0, 0, 0, 0};
  __builtin_amdgcn_tensor_load_to_lds(g0, g1, z4, z4, z8, 0);
#else
  __builtin_amdgcn_tensor_load_to_lds(g0, g1, z4, z4, 0);
#endif
}

__global__ void cvt_f32_to_bf16(const float* __restrict__ in,
                                bf16* __restrict__ out, int n) {
  int i = blockIdx.x * blockDim.x + threadIdx.x;
  if (i < n) out[i] = f2bf(in[i]);
}

// C[M,N] = A[M,K] * W[N,K]^T ; A,W bf16 row-major; out bf16 or f32.
// Block: 256 threads (8 waves), tile 128(M) x 64(N), KSTEP=64, double-buffered
// LDS ping-pong so global-load latency hides under the 8 WMMAs per stage.
__global__ __launch_bounds__(256) void gemm_bf16_wmma(
    const bf16* __restrict__ A, const bf16* __restrict__ W,
    bf16* __restrict__ Cb, float* __restrict__ Cf,
    int M, int N, int K, int write_f32) {
  __shared__ __align__(16) bf16 At[2][128 * 64];  // 2 x 16 KB
  __shared__ __align__(16) bf16 Wt[2][64 * 64];   // 2 x  8 KB
  const int tid = threadIdx.x;
  const int wave = tid >> 5, lane = tid & 31;
  const int bm = blockIdx.y * 128, bn = blockIdx.x * 64;

  const int arow = tid >> 1, acol = (tid & 1) * 32;
  const int wrow = tid >> 2, wcol = (tid & 3) * 16;
  const bf16* gA = A + (size_t)(bm + arow) * K + acol;
  const bf16* gW = W + (size_t)(bn + wrow) * K + wcol;
  bf16* lA0 = &At[0][arow * 64 + acol];
  bf16* lA1 = &At[1][arow * 64 + acol];
  bf16* lW0 = &Wt[0][wrow * 64 + wcol];
  bf16* lW1 = &Wt[1][wrow * 64 + wcol];

  const int nst = K / 64;
  bf16x8 ar[4], wr[2];
#pragma unroll
  for (int u = 0; u < 4; ++u) ar[u] = *(const bf16x8*)(gA + u * 8);
  wr[0] = *(const bf16x8*)(gW);
  wr[1] = *(const bf16x8*)(gW + 8);
#pragma unroll
  for (int u = 0; u < 4; ++u) *(bf16x8*)(lA0 + u * 8) = ar[u];
  *(bf16x8*)(lW0)     = wr[0];
  *(bf16x8*)(lW1 - (lW1 - lW0) + 8) = wr[1];  // = lW0 + 8

  f32x8 acc[4] = {};
  for (int s = 0; s < nst; ++s) {
    __syncthreads();                       // buf (s&1) fully stored + visible
    const int cur = s & 1;
    if (s + 1 < nst) {                     // stage s+1 global -> regs
      const bf16* nA = gA + (size_t)(s + 1) * 64;
      const bf16* nW = gW + (size_t)(s + 1) * 64;
#pragma unroll
      for (int u = 0; u < 4; ++u) ar[u] = *(const bf16x8*)(nA + u * 8);
      wr[0] = *(const bf16x8*)(nW);
      wr[1] = *(const bf16x8*)(nW + 8);
      if (s + 2 < nst) {                   // warm L2 for stage s+2
        __builtin_prefetch(gA + (size_t)(s + 2) * 64, 0, 1);
        __builtin_prefetch(gW + (size_t)(s + 2) * 64, 0, 1);
      }
    }
    // compute stage s out of buf[cur]
    bf16x16 a0 = load_a16x32(&At[cur][wave * 16 * 64], 64, lane);
    bf16x16 a1 = load_a16x32(&At[cur][wave * 16 * 64 + 32], 64, lane);
    bf16x16 bb[8];
#pragma unroll
    for (int nt = 0; nt < 4; ++nt) {
      bb[nt]     = load_b32x16(&Wt[cur][nt * 16 * 64], 64, lane);
      bb[4 + nt] = load_b32x16(&Wt[cur][nt * 16 * 64 + 32], 64, lane);
    }
#pragma unroll
    for (int nt = 0; nt < 4; ++nt) {
      acc[nt] = wmma_bf16(a0, bb[nt], acc[nt]);
      acc[nt] = wmma_bf16(a1, bb[4 + nt], acc[nt]);
    }
    if (s + 1 < nst) {                     // regs -> buf[cur^1]
      bf16* dA = cur ? lA0 : lA1;
      bf16* dW = cur ? lW0 : lW1;
#pragma unroll
      for (int u = 0; u < 4; ++u) *(bf16x8*)(dA + u * 8) = ar[u];
      *(bf16x8*)(dW)     = wr[0];
      *(bf16x8*)(dW + 8) = wr[1];
    }
  }
  const int mh = lane >> 4, n = lane & 15;
#pragma unroll
  for (int nt = 0; nt < 4; ++nt) {
#pragma unroll
    for (int i = 0; i < 8; ++i) {
      int row = bm + wave * 16 + i + 8 * mh;
      int col = bn + nt * 16 + n;
      if (write_f32) Cf[(size_t)row * N + col] = acc[nt][i];
      else           Cb[(size_t)row * N + col] = f2bf(acc[nt][i]);
    }
  }
}

// qkv raw [BS, 6144] bf16 -> RoPE(q,k) + scatter: Q,K [B*H,S,HD], V^T [B*H,HD,S]
__global__ void rope_scatter(const bf16* __restrict__ qkv,
                             const float* __restrict__ cosb,
                             const float* __restrict__ sinb,
                             bf16* __restrict__ Qo, bf16* __restrict__ Ko,
                             bf16* __restrict__ VTo) {
  int idx = blockIdx.x * blockDim.x + threadIdx.x;
  int j    = idx & 63;
  int part = (idx >> 6) % 3;
  int h    = (idx / 192) & (H - 1);
  int s    = (idx / (192 * H)) & (S - 1);
  int b    = idx / (192 * H * S);
  if (b >= Bz) return;
  const size_t rowbase = (size_t)(b * S + s) * NQKV;
  int f1 = h * (3 * HD) + part * HD + j;
  float x1 = bf2f(qkv[rowbase + f1]);
  float x2 = bf2f(qkv[rowbase + f1 + 64]);
  int bh = b * H + h;
  if (part < 2) {
    float c1 = cosb[s * HD + j],      sv1 = sinb[s * HD + j];
    float c2 = cosb[s * HD + j + 64], sv2 = sinb[s * HD + j + 64];
    float o1 = x1 * c1 - x2 * sv1;   // rotate_half: j<64 partner is -x[j+64]
    float o2 = x2 * c2 + x1 * sv2;
    bf16* dst = (part == 0) ? Qo : Ko;
    size_t base = ((size_t)bh * S + s) * HD;
    dst[base + j]      = f2bf(o1);
    dst[base + j + 64] = f2bf(o2);
  } else {
    VTo[((size_t)bh * HD + j) * S + s]      = f2bf(x1);
    VTo[((size_t)bh * HD + j + 64) * S + s] = f2bf(x2);
  }
}

// Causal flash attention. Block: 8 waves, 128 query rows (16/wave), key step 32.
// K/V tiles streamed by the Tensor Data Mover into ping-pong LDS buffers; the
// next step's DMA flies under the current step's WMMAs (wait tensorcnt<=2).
__global__ __launch_bounds__(256) void flash_attn_wmma(
    const bf16* __restrict__ Q, const bf16* __restrict__ Kc,
    const bf16* __restrict__ VT, bf16* __restrict__ O) {
  __shared__ __align__(16) bf16 Kt[2][32 * 128];  // [buf][key][hd]  16 KB
  __shared__ __align__(16) bf16 Vt[2][128 * 32];  // [buf][hd][key]  16 KB
  __shared__ __align__(16) bf16 Pt[8][16 * 32];   // per-wave P staging 8 KB
  const int tid = threadIdx.x;
  const int wave = tid >> 5, lane = tid & 31;
  const int n = lane & 15, hv = lane >> 4;
  const int bh = blockIdx.y;
  const int b = bh >> 4, h = bh & 15;
  const int qblock = blockIdx.x * 128;
  const int qbase  = qblock + wave * 16;
  const size_t bhS = (size_t)bh * S;
  const uint32_t kt_lds[2] = {(uint32_t)(uintptr_t)(void*)&Kt[0][0],
                              (uint32_t)(uintptr_t)(void*)&Kt[1][0]};
  const uint32_t vt_lds[2] = {(uint32_t)(uintptr_t)(void*)&Vt[0][0],
                              (uint32_t)(uintptr_t)(void*)&Vt[1][0]};

  bf16x16 aq[4];
#pragma unroll
  for (int c = 0; c < 4; ++c)
    aq[c] = load_a16x32(Q + (bhS + qbase) * HD + 32 * c, HD, lane);

  f32x8 o[8] = {};
  float mrow[8], lrow[8];
#pragma unroll
  for (int i = 0; i < 8; ++i) { mrow[i] = -__builtin_inff(); lrow[i] = 0.f; }

  const float scale = 0.08838834764831845f; // 1/sqrt(128)
  const int nsteps = qblock / 32 + 4;       // kb = 0..qblock+96

  if (wave == 0) {  // prologue: DMA step 0 into buf 0
    tdm_load_2d_b16(Kc + bhS * HD, kt_lds[0], HD, 32, HD, 32, HD);
    tdm_load_2d_b16(VT + (size_t)bh * HD * S, vt_lds[0], 32, HD, 32, HD, S);
  }

  for (int it = 0; it < nsteps; ++it) {
    const int kb = it * 32;
    const int cur = it & 1;
    if (wave == 0) {
      if (it + 1 < nsteps) {  // DMA step it+1 into the other buffer
        const int nb = (it + 1) & 1;
        tdm_load_2d_b16(Kc + (bhS + kb + 32) * HD, kt_lds[nb], HD, 32, HD, 32, HD);
        tdm_load_2d_b16(VT + (size_t)bh * HD * S + kb + 32, vt_lds[nb],
                        32, HD, 32, HD, S);
        __builtin_amdgcn_s_wait_tensorcnt(2);  // step it's pair has landed
      } else {
        __builtin_amdgcn_s_wait_tensorcnt(0);
      }
    }
    __syncthreads();  // tiles of step it visible to all waves
    if (kb <= qbase + 15) {
      f32x8 s0 = {}, s1 = {};
#pragma unroll
      for (int c = 0; c < 4; ++c) {
        bf16x16 b0 = load_b32x16(&Kt[cur][32 * c], 128, lane);
        bf16x16 b1 = load_b32x16(&Kt[cur][16 * 128 + 32 * c], 128, lane);
        s0 = wmma_bf16(aq[c], b0, s0);
        s1 = wmma_bf16(aq[c], b1, s1);
      }
      float corr[8];
#pragma unroll
      for (int i = 0; i < 8; ++i) {
        int qrow = qbase + i + 8 * hv;
        s0[i] = (kb + n      <= qrow) ? s0[i] * scale : -__builtin_inff();
        s1[i] = (kb + 16 + n <= qrow) ? s1[i] * scale : -__builtin_inff();
        float rm = fmaxf(s0[i], s1[i]);
#pragma unroll
        for (int mk = 1; mk < 16; mk <<= 1) rm = fmaxf(rm, __shfl_xor(rm, mk, 32));
        float mnew = fmaxf(mrow[i], rm);
        corr[i] = __expf(mrow[i] - mnew);
        mrow[i] = mnew;
        s0[i] = __expf(s0[i] - mnew);
        s1[i] = __expf(s1[i] - mnew);
        float rs = s0[i] + s1[i];
#pragma unroll
        for (int mk = 1; mk < 16; mk <<= 1) rs += __shfl_xor(rs, mk, 32);
        lrow[i] = lrow[i] * corr[i] + rs;
      }
#pragma unroll
      for (int t = 0; t < 8; ++t)
#pragma unroll
        for (int i = 0; i < 8; ++i) o[t][i] *= corr[i];

      // P: C-layout -> LDS -> A-layout (in-wave DS RAW, fenced with dscnt)
      bf16* P = &Pt[wave][0];
#pragma unroll
      for (int i = 0; i < 8; ++i) {
        int m = i + 8 * hv;
        P[m * 32 + n]      = f2bf(s0[i]);
        P[m * 32 + 16 + n] = f2bf(s1[i]);
      }
      asm volatile("s_wait_dscnt 0x0" ::: "memory");
      bf16x16 ap = load_a16x32(P, 32, lane);
#pragma unroll
      for (int t = 0; t < 8; ++t) {
        bf16x16 bv = load_b32x16(&Vt[cur][t * 16 * 32], 32, lane);
        o[t] = wmma_bf16(ap, bv, o[t]);
      }
    }
    __syncthreads();  // all waves done with buf[cur] before it is re-filled
  }
#pragma unroll
  for (int i = 0; i < 8; ++i) lrow[i] = 1.0f / lrow[i];
#pragma unroll
  for (int t = 0; t < 8; ++t)
#pragma unroll
    for (int i = 0; i < 8; ++i) {
      int qrow = qbase + i + 8 * hv;
      O[((size_t)b * S + qrow) * (H * HD) + h * HD + t * 16 + n] =
          f2bf(o[t][i] * lrow[i]);
    }
}

extern "C" void kernel_launch(void* const* d_in, const int* in_sizes, int n_in,
                              void* d_out, int out_size, void* d_ws, size_t ws_size,
                              hipStream_t stream) {
  (void)in_sizes; (void)n_in; (void)out_size; (void)ws_size;
  const float* hidden = (const float*)d_in[0];
  const float* cosb   = (const float*)d_in[1];
  const float* sinb   = (const float*)d_in[2];
  const float* w_qkv  = (const float*)d_in[3];
  const float* w_o    = (const float*)d_in[4];

  char* ws = (char*)d_ws;
  size_t off = 0;
  auto alloc = [&](size_t bytes) {
    char* p = ws + off;
    off += (bytes + 255) & ~(size_t)255;
    return p;
  };
  bf16* hid_bf  = (bf16*)alloc((size_t)BS * D * 2);          // reused as Q
  bf16* wqkv_bf = (bf16*)alloc((size_t)NQKV * D * 2);        // reused as K
  bf16* wo_bf   = (bf16*)alloc((size_t)D * (H * HD) * 2);
  bf16* qkv_raw = (bf16*)alloc((size_t)BS * NQKV * 2);       // reused as attn
  bf16* vt_bf   = (bf16*)alloc((size_t)Bz * H * HD * S * 2);

  bf16* Qb   = hid_bf;   // dead after GEMM1
  bf16* Kb   = wqkv_bf;  // dead after GEMM1
  bf16* attn = qkv_raw;  // dead after rope_scatter

  { int n = BS * D;
    cvt_f32_to_bf16<<<dim3((n + 255) / 256), dim3(256), 0, stream>>>(hidden, hid_bf, n); }
  { int n = NQKV * D;
    cvt_f32_to_bf16<<<dim3((n + 255) / 256), dim3(256), 0, stream>>>(w_qkv, wqkv_bf, n); }
  { int n = D * H * HD;
    cvt_f32_to_bf16<<<dim3((n + 255) / 256), dim3(256), 0, stream>>>(w_o, wo_bf, n); }

  gemm_bf16_wmma<<<dim3(NQKV / 64, BS / 128), dim3(256), 0, stream>>>(
      hid_bf, wqkv_bf, qkv_raw, nullptr, BS, NQKV, D, 0);

  { int n = Bz * S * H * 3 * 64;
    rope_scatter<<<dim3((n + 255) / 256), dim3(256), 0, stream>>>(
        qkv_raw, cosb, sinb, Qb, Kb, vt_bf); }

  flash_attn_wmma<<<dim3(S / 128, Bz * H), dim3(256), 0, stream>>>(Qb, Kb, vt_bf, attn);

  gemm_bf16_wmma<<<dim3(D / 64, BS / 128), dim3(256), 0, stream>>>(
      attn, wo_bf, nullptr, (float*)d_out, BS, D, H * HD, 1);
}
